// CrossAttention_58695023067822
// MI455X (gfx1250) — matrix operand order repaired and checked by reading, not compiled, not run on hardware
//
#include <hip/hip_runtime.h>
#include <hip/hip_bf16.h>

typedef __bf16 bf16_t;
typedef bf16_t v16bf __attribute__((ext_vector_type(16)));
typedef float  v8f   __attribute__((ext_vector_type(8)));
typedef unsigned int u32x4 __attribute__((ext_vector_type(4)));

union Frag { v16bf v; u32x4 u[2]; };

#define BATCH 4
#define CX    512
#define CI    64
#define NPIX  4096   // 64*64
#define CHUNKS (NPIX / 32)
#define PROW  40     // padded LDS row stride (bf16 elems) -> conflict-free

static __device__ __forceinline__ float xor16f(float v) {
  // lane i <-> lane i^16 swap via v_permlanex16_b32 (VALU, no DS wait)
  int i = __builtin_bit_cast(int, v);
  int r = __builtin_amdgcn_permlanex16(i, i, 0x76543210, 0xfedcba98, false, false);
  return __builtin_bit_cast(float, r);
}

static __device__ __forceinline__ v8f wmma_bf16(const Frag& a, const Frag& b, v8f c) {
  return __builtin_amdgcn_wmma_f32_16x16x32_bf16(false, a.v, false, b.v,
                                                 (short)0, c, false, false);
}

// ---------------------------------------------------------------------------
// Cast W matrices to bf16 (row-major kept). 327,680 elems total.
// ---------------------------------------------------------------------------
__global__ __launch_bounds__(256) void cast_w_kernel(
    const float* __restrict__ Wq, const float* __restrict__ Wk,
    const float* __restrict__ Wv, bf16_t* __restrict__ Wqb,
    bf16_t* __restrict__ Wkb, bf16_t* __restrict__ Wvb) {
  const int tid = blockIdx.x * 256 + threadIdx.x;
  if (tid < CI * CX)                 Wqb[tid]               = (bf16_t)Wq[tid];
  else if (tid < 2 * CI * CX)        Wkb[tid - CI * CX]     = (bf16_t)Wk[tid - CI * CX];
  else if (tid < 2 * CI * CX + CX * CX)
                                     Wvb[tid - 2 * CI * CX] = (bf16_t)Wv[tid - 2 * CI * CX];
}

// ---------------------------------------------------------------------------
// Cast + transpose X,Y [b][c][n] f32 -> Xt,Yt [b][n][c] bf16 via 64x64 LDS
// tiles (coalesced reads along n, coalesced writes along c).
// grid = (64, 8, 8): x=n-tile, y=c-tile, z = b + 4*isY. block = 256.
// ---------------------------------------------------------------------------
__global__ __launch_bounds__(256) void cast_tr_kernel(
    const float* __restrict__ X, const float* __restrict__ Y,
    bf16_t* __restrict__ Xt, bf16_t* __restrict__ Yt) {
  __shared__ bf16_t T[64 * 65];
  const int b   = blockIdx.z & 3;
  const int isY = blockIdx.z >> 2;
  const float* src = isY ? Y : X;
  bf16_t* dst      = isY ? Yt : Xt;
  const int n0 = blockIdx.x * 64;
  const int c0 = blockIdx.y * 64;

  for (int i = threadIdx.x; i < 4096; i += 256) {
    const int cl = i >> 6, nl = i & 63;
    T[cl * 65 + nl] = (bf16_t)src[((size_t)b * CX + c0 + cl) * NPIX + n0 + nl];
  }
  __syncthreads();
  for (int i = threadIdx.x; i < 4096; i += 256) {
    const int nl = i >> 6, cl = i & 63;
    dst[((size_t)b * NPIX + n0 + nl) * CX + c0 + cl] = T[cl * 65 + nl];
  }
}

// ---------------------------------------------------------------------------
// Q/K projection as WMMA GEMM: Qr[b][n][o] = sum_c Xt[b][n][c] W[o][c] + b[o]
// One wave per 16n x 64o tile, k = 512 in 16 chunks, 4 wmma per chunk.
// A fragments ping-pong one chunk ahead; B fragments batch-loaded per chunk.
// grid = (256, B, 2) block = 32.  z: 0=Q(Xt,Wq), 1=K(Yt,Wk).
// ---------------------------------------------------------------------------
__global__ __launch_bounds__(32) void projqk_wmma_kernel(
    const bf16_t* __restrict__ Xt, const bf16_t* __restrict__ Yt,
    const bf16_t* __restrict__ Wqb, const bf16_t* __restrict__ Wkb,
    const float* __restrict__ bq, const float* __restrict__ bk,
    bf16_t* __restrict__ Qr, bf16_t* __restrict__ Kr) {
  const int b   = blockIdx.y;
  const bool isK = (blockIdx.z != 0);
  const bf16_t* src  = isK ? Yt : Xt;
  const bf16_t* W    = isK ? Wkb : Wqb;
  const float*  bias = isK ? bk : bq;
  bf16_t* dst        = isK ? Kr : Qr;

  const int n0 = blockIdx.x * 16;
  const int lane = threadIdx.x, half = lane >> 4, lr = lane & 15;

  v8f acc[4] = {};
  const bf16_t* arow = src + ((size_t)b * NPIX + n0 + lr) * CX;
  Frag a[2];
  a[0].u[0] = *(const u32x4*)(arow + half * 8);
  a[0].u[1] = *(const u32x4*)(arow + 16 + half * 8);
  for (int kc = 0; kc < 16; ++kc) {
    Frag bf[4];
#pragma unroll
    for (int ot = 0; ot < 4; ++ot) {
      const bf16_t* wrow = W + ((size_t)(ot * 16 + lr)) * CX + kc * 32 + half * 8;
      bf[ot].u[0] = *(const u32x4*)(wrow);
      bf[ot].u[1] = *(const u32x4*)(wrow + 16);
    }
    if (kc + 1 < 16) {
      a[(kc + 1) & 1].u[0] = *(const u32x4*)(arow + (kc + 1) * 32 + half * 8);
      a[(kc + 1) & 1].u[1] = *(const u32x4*)(arow + (kc + 1) * 32 + 16 + half * 8);
    }
#pragma unroll
    for (int ot = 0; ot < 4; ++ot)
      acc[ot] = wmma_bf16(a[kc & 1], bf[ot], acc[ot]);
  }
#pragma unroll
  for (int ot = 0; ot < 4; ++ot) {
    const float bb = bias[ot * 16 + lr];
#pragma unroll
    for (int r = 0; r < 8; ++r)
      dst[((size_t)b * NPIX + n0 + r + 8 * half) * CI + ot * 16 + lr] =
          (bf16_t)(acc[ot][r] + bb);
  }
}

// ---------------------------------------------------------------------------
// V projection as WMMA GEMM: Vb[b][c][n] = sum_cy Wv[c][cy] Yt[b][n][cy] + bv[c]
// One wave per 16c x 128n tile: A (Wv rows) ping-ponged one chunk ahead,
// 8 B fragments batch-loaded per chunk and reused against one A.
// grid = (32, 32, B) block = 32.
// ---------------------------------------------------------------------------
__global__ __launch_bounds__(32) void projv_wmma_kernel(
    const bf16_t* __restrict__ Yt, const bf16_t* __restrict__ Wvb,
    const float* __restrict__ bv, bf16_t* __restrict__ Vb) {
  const int b  = blockIdx.z;
  const int cb = blockIdx.y * 16;
  const int n0 = blockIdx.x * 128;
  const int lane = threadIdx.x, half = lane >> 4, lr = lane & 15;

  v8f acc[8] = {};
  const bf16_t* arow = Wvb + ((size_t)(cb + lr)) * CX;
  Frag a[2];
  a[0].u[0] = *(const u32x4*)(arow + half * 8);
  a[0].u[1] = *(const u32x4*)(arow + 16 + half * 8);
  for (int kc = 0; kc < 16; ++kc) {
    Frag bf[8];
#pragma unroll
    for (int nt = 0; nt < 8; ++nt) {
      const bf16_t* yrow =
          Yt + ((size_t)b * NPIX + n0 + nt * 16 + lr) * CX + kc * 32 + half * 8;
      bf[nt].u[0] = *(const u32x4*)(yrow);
      bf[nt].u[1] = *(const u32x4*)(yrow + 16);
    }
    if (kc + 1 < 16) {
      a[(kc + 1) & 1].u[0] = *(const u32x4*)(arow + (kc + 1) * 32 + half * 8);
      a[(kc + 1) & 1].u[1] = *(const u32x4*)(arow + (kc + 1) * 32 + 16 + half * 8);
    }
#pragma unroll
    for (int nt = 0; nt < 8; ++nt)
      acc[nt] = wmma_bf16(a[kc & 1], bf[nt], acc[nt]);
  }
  float bb[8];
#pragma unroll
  for (int r = 0; r < 8; ++r) bb[r] = bv[cb + r + 8 * half];
#pragma unroll
  for (int nt = 0; nt < 8; ++nt)
#pragma unroll
    for (int r = 0; r < 8; ++r)
      Vb[((size_t)b * CX + cb + r + 8 * half) * NPIX + n0 + nt * 16 + lr] =
          (bf16_t)(acc[nt][r] + bb[r]);
}

// ---------------------------------------------------------------------------
// Flash attention (transposed), one wave32 per block:
//   S~ = K . Q^T  (keys in-lane)          4 wmma/chunk
//   O^T += V . P  (coalesced epilogue)   16 wmma/chunk
// Softmax: in-lane max/sum tree + one v_permlanex16; ballot-guarded rescale.
// grid = ((N/16)*2, B), block = 32.
// ---------------------------------------------------------------------------
__global__ __launch_bounds__(32) void attn_kernel(
    const float* __restrict__ X, const float* __restrict__ gamma,
    const bf16_t* __restrict__ Qr, const bf16_t* __restrict__ Kr,
    const bf16_t* __restrict__ Vb, float* __restrict__ out) {
  __shared__ alignas(16) bf16_t Plds[16 * PROW];

  const int b     = blockIdx.y;
  const int tile  = blockIdx.x >> 1;
  const int chalf = blockIdx.x & 1;
  const int n0 = tile * 16;
  const int c0 = chalf * 256;
  const int lane = threadIdx.x;
  const int half = lane >> 4;
  const int lr   = lane & 15;

  // Q as B-matrix fragments (cols n = n0+lr)
  Frag qb0, qb1;
  {
    const bf16_t* qrow = Qr + ((size_t)b * NPIX + n0 + lr) * CI;
    qb0.u[0] = *(const u32x4*)(qrow + half * 8);
    qb0.u[1] = *(const u32x4*)(qrow + half * 8 + 16);
    qb1.u[0] = *(const u32x4*)(qrow + 32 + half * 8);
    qb1.u[1] = *(const u32x4*)(qrow + 32 + half * 8 + 16);
  }

  // K as A-matrix fragments, refilled one chunk ahead
  Frag kf[2][2];
#pragma unroll
  for (int t = 0; t < 2; ++t) {
    const bf16_t* krow = Kr + ((size_t)b * NPIX + t * 16 + lr) * CI;
    kf[t][0].u[0] = *(const u32x4*)(krow + half * 8);
    kf[t][0].u[1] = *(const u32x4*)(krow + 16 + half * 8);
    kf[t][1].u[0] = *(const u32x4*)(krow + 32 + half * 8);
    kf[t][1].u[1] = *(const u32x4*)(krow + 32 + 16 + half * 8);
  }

  v8f acc[16] = {};                  // O^T accum: (c_local = r+8*half, n = lr)
  float mrun = -1e30f, lrun = 0.f;

  for (int j = 0; j < CHUNKS; ++j) {
    const int m0 = j * 32;

    v8f st[2];
#pragma unroll
    for (int t = 0; t < 2; ++t) {
      v8f s = {};
      s = wmma_bf16(kf[t][0], qb0, s);
      s = wmma_bf16(kf[t][1], qb1, s);
      st[t] = s;
    }

    // issue next chunk's K loads; softmax + PV hide the latency
    if (j + 1 < CHUNKS) {
#pragma unroll
      for (int t = 0; t < 2; ++t) {
        const bf16_t* krow =
            Kr + ((size_t)b * NPIX + m0 + 32 + t * 16 + lr) * CI;
        kf[t][0].u[0] = *(const u32x4*)(krow + half * 8);
        kf[t][0].u[1] = *(const u32x4*)(krow + 16 + half * 8);
        kf[t][1].u[0] = *(const u32x4*)(krow + 32 + half * 8);
        kf[t][1].u[1] = *(const u32x4*)(krow + 32 + 16 + half * 8);
      }
    }

    // online softmax: 16 key-scores per lane + one xor-16 permlane
    float mx = st[0][0];
#pragma unroll
    for (int r = 1; r < 8; ++r) mx = fmaxf(mx, st[0][r]);
#pragma unroll
    for (int r = 0; r < 8; ++r) mx = fmaxf(mx, st[1][r]);
    mx = fmaxf(mx, xor16f(mx));
    const float mnew = fmaxf(mrun, mx);
    const float sc   = __expf(mrun - mnew);
    float ps = 0.f;
#pragma unroll
    for (int t = 0; t < 2; ++t)
#pragma unroll
      for (int r = 0; r < 8; ++r) {
        const float p = __expf(st[t][r] - mnew);
        ps += p;
        Plds[lr * PROW + r + 8 * half + t * 16] = (bf16_t)p;   // P[n][m]
      }
    ps += xor16f(ps);
    lrun = lrun * sc + ps;
    const bool grew = mnew > mrun;
    mrun = mnew;
    if (__ballot(grew)) {            // uniform branch: rescale only when needed
#pragma unroll
      for (int ct = 0; ct < 16; ++ct) acc[ct] *= sc;
    }

    asm volatile("s_wait_dscnt 0" ::: "memory");

    Frag pb;                         // P as B-matrix fragment (32m x 16n)
    pb.u[0] = *(const u32x4*)(&Plds[lr * PROW + half * 8]);
    pb.u[1] = *(const u32x4*)(&Plds[lr * PROW + half * 8 + 16]);

    Frag vf[2];                      // V fragments ping-ponged
    {
      const bf16_t* vrow = Vb + ((size_t)b * CX + c0 + lr) * NPIX + m0;
      vf[0].u[0] = *(const u32x4*)(vrow + half * 8);
      vf[0].u[1] = *(const u32x4*)(vrow + 16 + half * 8);
    }
#pragma unroll
    for (int ct = 0; ct < 16; ++ct) {
      if (ct + 1 < 16) {
        const bf16_t* vrow =
            Vb + ((size_t)b * CX + c0 + (ct + 1) * 16 + lr) * NPIX + m0;
        vf[(ct + 1) & 1].u[0] = *(const u32x4*)(vrow + half * 8);
        vf[(ct + 1) & 1].u[1] = *(const u32x4*)(vrow + 16 + half * 8);
      }
      acc[ct] = wmma_bf16(vf[ct & 1], pb, acc[ct]);
    }
  }

  // epilogue: out = gamma*O/l + X, coalesced 64B row stores
  const float g = gamma[0];
  const float invl = g / lrun;
#pragma unroll
  for (int ct = 0; ct < 16; ++ct) {
#pragma unroll
    for (int r = 0; r < 8; ++r) {
      const int c = c0 + ct * 16 + r + 8 * half;
      const size_t row = ((size_t)b * CX + c) * NPIX + n0;
      out[row + lr] = acc[ct][r] * invl + X[row + lr];
    }
  }
}

// ---------------------------------------------------------------------------
extern "C" void kernel_launch(void* const* d_in, const int* in_sizes, int n_in,
                              void* d_out, int out_size, void* d_ws, size_t ws_size,
                              hipStream_t stream) {
  const float* X     = (const float*)d_in[0];
  const float* Y     = (const float*)d_in[1];
  const float* Wq    = (const float*)d_in[2];
  const float* bq    = (const float*)d_in[3];
  const float* Wk    = (const float*)d_in[4];
  const float* bk    = (const float*)d_in[5];
  const float* Wv    = (const float*)d_in[6];
  const float* bv    = (const float*)d_in[7];
  const float* gamma = (const float*)d_in[8];
  float* out = (float*)d_out;

  // ws: Xt 16.8MB | Yt 16.8MB | Qr 2MB | Kr 2MB | Wq/Wk/Wv bf16 0.65MB
  bf16_t* Xt  = (bf16_t*)d_ws;
  bf16_t* Yt  = Xt  + (size_t)BATCH * NPIX * CX;
  bf16_t* Qr  = Yt  + (size_t)BATCH * NPIX * CX;
  bf16_t* Kr  = Qr  + (size_t)BATCH * NPIX * CI;
  bf16_t* Wqb = Kr  + (size_t)BATCH * NPIX * CI;
  bf16_t* Wkb = Wqb + (size_t)CI * CX;
  bf16_t* Wvb = Wkb + (size_t)CI * CX;
  bf16_t* Vb  = Xt;   // alias: Xt is dead after Q-proj (stream-ordered)

  cast_w_kernel<<<dim3((2 * CI * CX + CX * CX + 255) / 256), 256, 0, stream>>>(
      Wq, Wk, Wv, Wqb, Wkb, Wvb);
  cast_tr_kernel<<<dim3(NPIX / 64, CX / 64, 2 * BATCH), 256, 0, stream>>>(
      X, Y, Xt, Yt);
  projqk_wmma_kernel<<<dim3(NPIX / 16, BATCH, 2), 32, 0, stream>>>(
      Xt, Yt, Wqb, Wkb, bq, bk, Qr, Kr);
  projv_wmma_kernel<<<dim3(NPIX / 128, CX / 16, BATCH), 32, 0, stream>>>(
      Yt, Wvb, bv, Vb);
  attn_kernel<<<dim3((NPIX / 16) * 2, BATCH), 32, 0, stream>>>(
      X, gamma, Qr, Kr, Vb, out);
}
